// MultiHeadedAttentionBlock_71116068487276
// MI455X (gfx1250) — matrix operand order repaired and checked by reading, not compiled, not run on hardware
//
#include <hip/hip_runtime.h>

// ---------------------------------------------------------------------------
// MI455X (gfx1250) fused causal MHA block.
//   k1: QKV = x @ W_kqv + b  -> f16 workspace, head-major [c][b][h][t][dk]
//   k2: flash attention (causal, online softmax) -> f16 [b][t][d]
//   k3: out = attn @ W_o + b_o -> f32 d_out
// Matmuls via v_wmma_f32_16x16x32_f16 (wave32 WMMA, f32 accumulate).
// GEMMs: double-buffered LDS (global loads overlap WMMA inside barrier window).
// Attention: Q/K tiles staged with the CDNA5 async copy engine
// (global_load_async_to_lds_b128 / ASYNCcnt, inline asm per ISA ch.8 sec.4),
// global_prefetch for next tiles, wave-private s_wait_dscnt P exchange.
// Workspace: 3*CHUNK + B*T*D halves = 32 MB.
// ---------------------------------------------------------------------------

typedef _Float16 half16 __attribute__((ext_vector_type(16)));
typedef _Float16 half8  __attribute__((ext_vector_type(8)));
typedef float    float8 __attribute__((ext_vector_type(8)));

#define BATCH  2
#define TSEQ   2048
#define DMODEL 1024
#define HEADS  16
#define DK     64
#define BH     (BATCH * HEADS)
#define CHUNK  ((size_t)BH * TSEQ * DK)   // elements per K/Q/V chunk

// ---- CDNA5 async global -> LDS copy (per-lane 16B, tracked by ASYNCcnt) ----
// VDST = lane's LDS byte address (low 32 bits of a generic LDS pointer),
// VADDR = lane's 64-bit global address. Completion: s_wait_asynccnt 0.
__device__ __forceinline__ void async_copy_b128(_Float16* lds_dst,
                                                const _Float16* gsrc) {
  const unsigned ldsoff = (unsigned)(uintptr_t)lds_dst;       // LDS byte offset
  const unsigned long long ga = (unsigned long long)(uintptr_t)gsrc;
  asm volatile("global_load_async_to_lds_b128 %0, %1, off"
               :: "v"(ldsoff), "v"(ga) : "memory");
}

__device__ __forceinline__ void async_wait_all() {
  asm volatile("s_wait_asynccnt 0x0" ::: "memory");
}

__device__ __forceinline__ float8 zero8() {
  float8 z;
#pragma unroll
  for (int i = 0; i < 8; ++i) z[i] = 0.0f;
  return z;
}

// B-fragment: 16 contiguous halves (lane = column, halves = K run).
__device__ __forceinline__ half16 ld_frag_b16(const _Float16* p) {
  const half8* q = (const half8*)p;
  half8 a = q[0], b = q[1];
  half16 r;
#pragma unroll
  for (int i = 0; i < 8; ++i) { r[i] = a[i]; r[8 + i] = b[i]; }
  return r;
}

// A-fragment: lanes 0-15 hold K{0-15}, lanes 16-31 hold K{16-31}.
__device__ __forceinline__ half16 ld_frag_a16(const _Float16* rowp, int hi) {
  half8 a = *(const half8*)(rowp + 2 * hi);
  half8 b = *(const half8*)(rowp + 2 * hi + 8);
  half16 r;
#pragma unroll
  for (int i = 0; i < 8; ++i) { r[i] = a[i]; r[8 + i] = b[i]; }
  return r;
}

__device__ __forceinline__ float8 wmma16(half16 a, half16 b, float8 c) {
  return __builtin_amdgcn_wmma_f32_16x16x32_f16(false, a, false, b,
                                                (short)0, c, false, false);
}

// ---- GEMM tile staging helpers (-> f16 LDS) --------------------------------
__device__ __forceinline__ void stage_a_f32(_Float16* Asb, const float* __restrict__ x,
                                            int bm, int k0, int tid) {
#pragma unroll
  for (int i = 0; i < 4; ++i) {
    int e = (tid + i * 256) * 4;
    int r = e >> 5, kk = e & 31;
    float4 v = *(const float4*)(x + (size_t)(bm + r) * DMODEL + k0 + kk);
    _Float16* d = Asb + r * 32 + kk;
    d[0] = (_Float16)v.x; d[1] = (_Float16)v.y;
    d[2] = (_Float16)v.z; d[3] = (_Float16)v.w;
  }
}

__device__ __forceinline__ void stage_a_f16(_Float16* Asb, const _Float16* __restrict__ A,
                                            int bm, int k0, int tid) {
#pragma unroll
  for (int i = 0; i < 2; ++i) {
    int e = tid * 8 + i * 2048;
    int r = e >> 5, kk = e & 31;
    *(half8*)(Asb + r * 32 + kk) =
        *(const half8*)(A + (size_t)(bm + r) * DMODEL + k0 + kk);
  }
}

__device__ __forceinline__ void stage_b_f32(_Float16* Bsb, const float* __restrict__ W,
                                            int ldw, int bn, int k0, int tid) {
#pragma unroll
  for (int i = 0; i < 4; ++i) {
    int e = (tid + i * 256) * 4;
    int kk = e >> 7, c = e & 127;
    float4 v = *(const float4*)(W + (size_t)(k0 + kk) * ldw + bn + c);
    Bsb[(c + 0) * 40 + kk] = (_Float16)v.x;
    Bsb[(c + 1) * 40 + kk] = (_Float16)v.y;
    Bsb[(c + 2) * 40 + kk] = (_Float16)v.z;
    Bsb[(c + 3) * 40 + kk] = (_Float16)v.w;
  }
}

// ---------------------------------------------------------------------------
// Kernel 1: QKV projection. M=4096, K=1024, N=3072. 128x128x32 tiles,
// double-buffered LDS. Writes f16 kqv[c][b][h][t][dk] (split order K,Q,V).
// ---------------------------------------------------------------------------
__global__ __launch_bounds__(256) void qkv_gemm_kernel(
    const float* __restrict__ x, const float* __restrict__ Wkqv,
    const float* __restrict__ bkqv, _Float16* __restrict__ kqv) {
  __shared__ _Float16 As[2][128 * 32];   // [row][k]
  __shared__ _Float16 Bs[2][128 * 40];   // [col][k], padded stride 40

  const int tid  = threadIdx.x;
  const int lane = tid & 31;
  const int w    = tid >> 5;
  const int wm   = w & 3;
  const int wn   = w >> 2;
  const int l16  = lane & 15;
  const int halfId = (lane >> 4) & 1;
  const int hiA  = halfId ? 8 : 0;
  const int hiB  = halfId ? 16 : 0;
  const int bm   = blockIdx.x * 128;
  const int bn   = blockIdx.y * 128;

  float8 acc[2][4];
#pragma unroll
  for (int mf = 0; mf < 2; ++mf)
#pragma unroll
    for (int nf = 0; nf < 4; ++nf) acc[mf][nf] = zero8();

  stage_a_f32(As[0], x, bm, 0, tid);
  stage_b_f32(Bs[0], Wkqv, 3072, bn, 0, tid);
  __syncthreads();

  for (int k0 = 0; k0 < DMODEL; k0 += 32) {
    const int cur = (k0 >> 5) & 1;
    if (k0 + 32 < DMODEL) {            // prefetch next tile into other buffer
      stage_a_f32(As[cur ^ 1], x, bm, k0 + 32, tid);
      stage_b_f32(Bs[cur ^ 1], Wkqv, 3072, bn, k0 + 32, tid);
    }
    half16 af[2], bf[4];
#pragma unroll
    for (int mf = 0; mf < 2; ++mf)
      af[mf] = ld_frag_a16(As[cur] + (wm * 32 + mf * 16 + l16) * 32, hiA);
#pragma unroll
    for (int nf = 0; nf < 4; ++nf)
      bf[nf] = ld_frag_b16(Bs[cur] + (wn * 64 + nf * 16 + l16) * 40 + hiB);
#pragma unroll
    for (int mf = 0; mf < 2; ++mf)
#pragma unroll
      for (int nf = 0; nf < 4; ++nf)
        acc[mf][nf] = wmma16(af[mf], bf[nf], acc[mf][nf]);
    __syncthreads();
  }

  // Epilogue: add bias, scatter f16 to head-major layout.
#pragma unroll
  for (int mf = 0; mf < 2; ++mf) {
#pragma unroll
    for (int nf = 0; nf < 4; ++nf) {
      const int col  = bn + wn * 64 + nf * 16 + l16;     // [0,3072)
      const float bias = bkqv[col];
      const int c    = col >> 10;                        // 0=K,1=Q,2=V
      const int rem  = col & 1023;
      const int h    = rem >> 6, dd = rem & 63;
#pragma unroll
      for (int r = 0; r < 8; ++r) {
        const int row = bm + wm * 32 + mf * 16 + r + 8 * halfId;
        const int b   = row >> 11, t = row & 2047;
        kqv[(((size_t)c * BATCH + b) * HEADS + h) * TSEQ * DK + (size_t)t * DK + dd] =
            (_Float16)(acc[mf][nf][r] + bias);
      }
    }
  }
}

// ---------------------------------------------------------------------------
// Kernel 2: flash attention, causal. Block = (b,h,128-row q tile), 8 waves,
// each wave owns a 16-row q strip; 32 keys per inner step. Q/K tiles staged
// with the async copy engine (ASYNCcnt), V transposed manually.
// ---------------------------------------------------------------------------
__global__ __launch_bounds__(256) void flash_attn_kernel(
    const _Float16* __restrict__ Qp, const _Float16* __restrict__ Kp,
    const _Float16* __restrict__ Vp, _Float16* __restrict__ attn) {
  __shared__ _Float16 Qs[128 * DK];     // [qrow][dk]   16 KB
  __shared__ _Float16 Ks[32 * DK];      // [key][dk]     4 KB
  __shared__ _Float16 Vts[DK * 32];     // [dk][key]     4 KB (transposed)
  __shared__ _Float16 Ps[8 * 16 * 32];  // per-wave P    8 KB

  const int tid    = threadIdx.x;
  const int lane   = tid & 31;
  const int w      = tid >> 5;
  const int l16    = lane & 15;
  const int halfId = (lane >> 4) & 1;
  const int hiA    = halfId ? 8 : 0;
  const int hiB    = halfId ? 16 : 0;

  const int qt = blockIdx.x;            // 0..15 query tile
  const int bh = blockIdx.y;            // 0..31
  const int b  = bh >> 4, h = bh & 15;

  const _Float16* Qb = Qp + (size_t)bh * TSEQ * DK;
  const _Float16* Kb = Kp + (size_t)bh * TSEQ * DK;
  const _Float16* Vb = Vp + (size_t)bh * TSEQ * DK;

  // Stage Q tile (contiguous 128*64 halves) via async copy engine.
  {
    const _Float16* src = Qb + (size_t)qt * 128 * DK;
#pragma unroll
    for (int i = 0; i < 4; ++i)
      async_copy_b128(Qs + (tid + i * 256) * 8, src + (size_t)(tid + i * 256) * 8);
    async_wait_all();
  }
  __syncthreads();

  const half16 qa0 = ld_frag_a16(Qs + (w * 16 + l16) * DK + 0, hiA);
  const half16 qa1 = ld_frag_a16(Qs + (w * 16 + l16) * DK + 32, hiA);

  float8 o0 = zero8(), o1 = zero8(), o2 = zero8(), o3 = zero8();
  float m[8], l[8];
#pragma unroll
  for (int r = 0; r < 8; ++r) { m[r] = -3.0e38f; l[r] = 0.0f; }

  const int qrow0 = qt * 128 + w * 16;  // wave's first absolute q row
  const int nkt   = (qt + 1) * 4;       // causal: key tiles 0..nkt-1

  for (int kt = 0; kt < nkt; ++kt) {
    const int kb0 = kt * 32;
    __syncthreads();                    // prior compute done before restaging
    // Stage K tile (contiguous 2048 halves) via async copy engine.
    async_copy_b128(Ks + tid * 8, Kb + (size_t)kb0 * DK + (size_t)tid * 8);
    // Stage V tile transposed: Vts[dk][key]
    {
      const int key = tid >> 3;
      const int dk0 = (tid & 7) * 8;
      half8 v = *(const half8*)(Vb + (size_t)(kb0 + key) * DK + dk0);
#pragma unroll
      for (int j = 0; j < 8; ++j) Vts[(dk0 + j) * 32 + key] = v[j];
    }
    // Prefetch next step's K/V into cache while this step computes.
    if (kt + 1 < nkt) {
      __builtin_prefetch(Kb + (size_t)(kb0 + 32) * DK, 0, 1);
      __builtin_prefetch(Vb + (size_t)(kb0 + 32) * DK, 0, 1);
    }
    async_wait_all();                   // K tile resident in LDS
    __syncthreads();

    if (kb0 <= qrow0 + 15) {
      // S = Q K^T, two 16x16 tiles (keys kb0..+15, kb0+16..+31)
      float8 s0 = zero8(), s1 = zero8();
      s0 = wmma16(qa0, ld_frag_b16(Ks + l16 * DK + hiB), s0);
      s0 = wmma16(qa1, ld_frag_b16(Ks + l16 * DK + 32 + hiB), s0);
      s1 = wmma16(qa0, ld_frag_b16(Ks + (16 + l16) * DK + hiB), s1);
      s1 = wmma16(qa1, ld_frag_b16(Ks + (16 + l16) * DK + 32 + hiB), s1);

      // scale 1/sqrt(dk) + causal mask
#pragma unroll
      for (int r = 0; r < 8; ++r) {
        const int row = qrow0 + r + 8 * halfId;
        s0[r] = (kb0 + l16      <= row) ? s0[r] * 0.125f : -3.0e38f;
        s1[r] = (kb0 + 16 + l16 <= row) ? s1[r] * 0.125f : -3.0e38f;
      }

      // online softmax (rows live in 16-lane halves of the wave)
      float fs[8];
#pragma unroll
      for (int r = 0; r < 8; ++r) {
        float mx = fmaxf(s0[r], s1[r]);
#pragma unroll
        for (int d = 1; d < 16; d <<= 1) mx = fmaxf(mx, __shfl_xor(mx, d, 32));
        const float mnew = fmaxf(m[r], mx);
        fs[r] = expf(m[r] - mnew);
        m[r]  = mnew;
        s0[r] = expf(s0[r] - mnew);
        s1[r] = expf(s1[r] - mnew);
        float rs = s0[r] + s1[r];
#pragma unroll
        for (int d = 1; d < 16; d <<= 1) rs += __shfl_xor(rs, d, 32);
        l[r] = l[r] * fs[r] + rs;
      }
#pragma unroll
      for (int r = 0; r < 8; ++r) {
        o0[r] *= fs[r]; o1[r] *= fs[r]; o2[r] *= fs[r]; o3[r] *= fs[r];
      }

      // P (C-layout) -> per-wave LDS -> A-layout fragment
      _Float16* pw = Ps + w * (16 * 32);
#pragma unroll
      for (int r = 0; r < 8; ++r) {
        pw[(r + 8 * halfId) * 32 + l16]      = (_Float16)s0[r];
        pw[(r + 8 * halfId) * 32 + 16 + l16] = (_Float16)s1[r];
      }
      asm volatile("s_wait_dscnt 0x0" ::: "memory");  // wave-private exchange
      const half16 pa = ld_frag_a16(pw + l16 * 32, hiA);

      o0 = wmma16(pa, ld_frag_b16(Vts + (0 * 16 + l16) * 32 + hiB), o0);
      o1 = wmma16(pa, ld_frag_b16(Vts + (1 * 16 + l16) * 32 + hiB), o1);
      o2 = wmma16(pa, ld_frag_b16(Vts + (2 * 16 + l16) * 32 + hiB), o2);
      o3 = wmma16(pa, ld_frag_b16(Vts + (3 * 16 + l16) * 32 + hiB), o3);
    }
  }

  // Normalize and store attn[b][t][h*64 + dk] (heads transposed back)
#pragma unroll
  for (int r = 0; r < 8; ++r) {
    const float inv = 1.0f / l[r];
    const int t = qrow0 + r + 8 * halfId;
    _Float16* dst = attn + ((size_t)b * TSEQ + t) * DMODEL + h * DK;
    dst[0 * 16 + l16] = (_Float16)(o0[r] * inv);
    dst[1 * 16 + l16] = (_Float16)(o1[r] * inv);
    dst[2 * 16 + l16] = (_Float16)(o2[r] * inv);
    dst[3 * 16 + l16] = (_Float16)(o3[r] * inv);
  }
}

// ---------------------------------------------------------------------------
// Kernel 3: out = attn @ W_o + b_o. M=4096, K=1024, N=1024. Double-buffered.
// ---------------------------------------------------------------------------
__global__ __launch_bounds__(256) void out_gemm_kernel(
    const _Float16* __restrict__ A, const float* __restrict__ Wo,
    const float* __restrict__ bo, float* __restrict__ out) {
  __shared__ _Float16 As[2][128 * 32];
  __shared__ _Float16 Bs[2][128 * 40];

  const int tid  = threadIdx.x;
  const int lane = tid & 31;
  const int w    = tid >> 5;
  const int wm   = w & 3;
  const int wn   = w >> 2;
  const int l16  = lane & 15;
  const int halfId = (lane >> 4) & 1;
  const int hiA  = halfId ? 8 : 0;
  const int hiB  = halfId ? 16 : 0;
  const int bm   = blockIdx.x * 128;
  const int bn   = blockIdx.y * 128;

  float8 acc[2][4];
#pragma unroll
  for (int mf = 0; mf < 2; ++mf)
#pragma unroll
    for (int nf = 0; nf < 4; ++nf) acc[mf][nf] = zero8();

  stage_a_f16(As[0], A, bm, 0, tid);
  stage_b_f32(Bs[0], Wo, DMODEL, bn, 0, tid);
  __syncthreads();

  for (int k0 = 0; k0 < DMODEL; k0 += 32) {
    const int cur = (k0 >> 5) & 1;
    if (k0 + 32 < DMODEL) {
      stage_a_f16(As[cur ^ 1], A, bm, k0 + 32, tid);
      stage_b_f32(Bs[cur ^ 1], Wo, DMODEL, bn, k0 + 32, tid);
    }
    half16 af[2], bf[4];
#pragma unroll
    for (int mf = 0; mf < 2; ++mf)
      af[mf] = ld_frag_a16(As[cur] + (wm * 32 + mf * 16 + l16) * 32, hiA);
#pragma unroll
    for (int nf = 0; nf < 4; ++nf)
      bf[nf] = ld_frag_b16(Bs[cur] + (wn * 64 + nf * 16 + l16) * 40 + hiB);
#pragma unroll
    for (int mf = 0; mf < 2; ++mf)
#pragma unroll
      for (int nf = 0; nf < 4; ++nf)
        acc[mf][nf] = wmma16(af[mf], bf[nf], acc[mf][nf]);
    __syncthreads();
  }

#pragma unroll
  for (int mf = 0; mf < 2; ++mf) {
#pragma unroll
    for (int nf = 0; nf < 4; ++nf) {
      const int col = bn + wn * 64 + nf * 16 + l16;
      const float bias = bo[col];
#pragma unroll
      for (int r = 0; r < 8; ++r) {
        const int row = bm + wm * 32 + mf * 16 + r + 8 * halfId;
        out[(size_t)row * DMODEL + col] = acc[mf][nf][r] + bias;
      }
    }
  }
}

// ---------------------------------------------------------------------------
extern "C" void kernel_launch(void* const* d_in, const int* in_sizes, int n_in,
                              void* d_out, int out_size, void* d_ws, size_t ws_size,
                              hipStream_t stream) {
  const float* x    = (const float*)d_in[0];
  const float* Wkqv = (const float*)d_in[1];
  const float* bkqv = (const float*)d_in[2];
  const float* Wo   = (const float*)d_in[3];
  const float* bo   = (const float*)d_in[4];
  float* out        = (float*)d_out;

  _Float16* kqv16  = (_Float16*)d_ws;                 // [3][b][h][t][dk], 24 MB
  _Float16* attn16 = kqv16 + 3 * CHUNK;               // [b][t][d], 8 MB

  // 1) QKV projection (M=4096, N=3072)
  qkv_gemm_kernel<<<dim3(4096 / 128, 3072 / 128), 256, 0, stream>>>(
      x, Wkqv, bkqv, kqv16);

  // 2) causal flash attention; split order is K,Q,V
  flash_attn_kernel<<<dim3(TSEQ / 128, BH), 256, 0, stream>>>(
      kqv16 + CHUNK /*Q*/, kqv16 /*K*/, kqv16 + 2 * CHUNK /*V*/, attn16);

  // 3) output projection (M=4096, N=1024)
  out_gemm_kernel<<<dim3(4096 / 128, DMODEL / 128), 256, 0, stream>>>(
      attn16, Wo, bo, out);
}
